// CostVolumeLayer_23897198035242
// MI455X (gfx1250) — matrix operand order repaired and checked by reading, not compiled, not run on hardware
//
#include <hip/hip_runtime.h>

typedef __attribute__((ext_vector_type(16))) __bf16 v16bf;
typedef __attribute__((ext_vector_type(8)))  float  v8f;

#define B_   8
#define C_   128
#define H_   160
#define W_   288
#define S_   4
#define ND_  81
#define HW_  (H_ * W_)
#define SCALE_ (1.0f / 128.0f)

// LDS layout: [pixel][channel], channel-contiguous bf16, padded stride
#define PIXSTRIDE 136                    // 128 ch + 8 bf16 pad (272 B, 16B aligned)
#define TGT_ROWS  12                     // y0-4 .. y0+7
#define TGT_PX    32                     // x0-4 .. x0+27
#define SRC_ROWS  4
#define SRC_PX    16
#define TGT_BYTES (TGT_ROWS * TGT_PX * PIXSTRIDE * 2)
#define SRC_BYTES (SRC_ROWS * SRC_PX * PIXSTRIDE * 2)

// Per-wave D-tile staging buffer: 32 rows (tgt px) x 16 cols (src px), padded
#define DSTR       18                    // f32 units; even -> 8B-aligned b64 stores
#define DBUF_WAVE  (32 * DSTR * 4)       // bytes per wave
#define DBUF_BYTES (4 * DBUF_WAVE)

union Frag { v16bf v; uint4 q[2]; };

__device__ __forceinline__ unsigned short f2bf(float f) {
    union { float f; unsigned u; } x; x.f = f;
    unsigned r = x.u + 0x7FFFu + ((x.u >> 16) & 1u);   // round-to-nearest-even
    return (unsigned short)(r >> 16);
}

// Exact channel ordering of the reference: ch0=(0,0); per i: (-i,0),(i,0),(0,-i),(0,i);
// then per j: (-i,-j),(i,j),(-i,j),(i,-j).
__device__ __forceinline__ int chidx(int dy, int dx) {
    if (dy == 0 && dx == 0) return 0;
    int ai = dy < 0 ? -dy : dy;
    int aj = dx < 0 ? -dx : dx;
    if (dx == 0) return 1 + (ai - 1) * 20 + (dy < 0 ? 0 : 1);
    if (dy == 0) return 1 + (aj - 1) * 20 + (dx < 0 ? 2 : 3);
    int q = (dy < 0) ? (dx < 0 ? 0 : 2) : (dx > 0 ? 1 : 3);
    return 1 + (ai - 1) * 20 + 4 + (aj - 1) * 4 + q;
}

extern "C" __global__ void __launch_bounds__(128)
costvol_wmma_bf16(const float* __restrict__ src, const float* __restrict__ tgt,
                  float* __restrict__ out) {
    extern __shared__ char smem[];
    char* smemT = smem;                               // tgt tile (bf16)
    char* smemS = smem + TGT_BYTES;                   // src tile (bf16, pre-scaled by 1/C)
    char* smemD = smem + TGT_BYTES + SRC_BYTES;       // per-wave f32 D tiles

    const int x0 = blockIdx.x * 16;
    const int y0 = blockIdx.y * 4;
    const int b  = blockIdx.z;
    const int t  = threadIdx.x;

    // ---- stage tgt: 12 rows x 16 px-pairs x 64 ch-pairs, f32 -> bf16, [px][ch] transpose.
    // x window starts at the even offset x0-4 and W is even, so each pixel pair is
    // either fully in range or fully out -> single predicate per float2 load.
    for (int task = t; task < TGT_ROWS * 64 * 16; task += 128) {
        int pp  = task & 15;             // pixel pair in window
        int c2  = (task >> 4) & 63;      // channel pair
        int row = task >> 10;            // / (16*64)
        int gy  = y0 - S_ + row;
        int gx  = x0 - S_ + 2 * pp;
        unsigned lo = 0u, hi = 0u;
        if ((unsigned)gy < (unsigned)H_ && (unsigned)gx < (unsigned)W_) {
            const float* p = tgt + (((size_t)b * C_ + 2 * c2) * H_ + gy) * W_ + gx;
            float2 ca = *(const float2*)p;          // channel 2*c2,   px gx, gx+1
            float2 cb = *(const float2*)(p + HW_);  // channel 2*c2+1, px gx, gx+1
            lo = (unsigned)f2bf(ca.x) | ((unsigned)f2bf(cb.x) << 16);
            hi = (unsigned)f2bf(ca.y) | ((unsigned)f2bf(cb.y) << 16);
        }
        char* q = smemT + ((row * TGT_PX + 2 * pp) * PIXSTRIDE + 2 * c2) * 2;
        *(unsigned*)q                   = lo;
        *(unsigned*)(q + PIXSTRIDE * 2) = hi;
    }
    // ---- stage src: 4 rows x 8 px-pairs x 64 ch-pairs (always in range), scaled by 1/C
    for (int task = t; task < SRC_ROWS * 64 * 8; task += 128) {
        int pp  = task & 7;
        int c2  = (task >> 3) & 63;
        int row = task >> 9;             // / (8*64)
        const float* p = src + (((size_t)b * C_ + 2 * c2) * H_ + (y0 + row)) * W_
                             + (x0 + 2 * pp);
        float2 ca = *(const float2*)p;
        float2 cb = *(const float2*)(p + HW_);
        unsigned lo = (unsigned)f2bf(ca.x * SCALE_) | ((unsigned)f2bf(cb.x * SCALE_) << 16);
        unsigned hi = (unsigned)f2bf(ca.y * SCALE_) | ((unsigned)f2bf(cb.y * SCALE_) << 16);
        char* q = smemS + ((row * SRC_PX + 2 * pp) * PIXSTRIDE + 2 * c2) * 2;
        *(unsigned*)q                   = lo;
        *(unsigned*)(q + PIXSTRIDE * 2) = hi;
    }
    __syncthreads();

    const int wv   = t >> 5;              // wave id = local y row
    const int lane = t & 31;
    const int lh   = lane >> 4;           // half-wave select
    const int l16  = lane & 15;
    const int y    = y0 + wv;
    float* dbuf = (float*)(smemD + wv * DBUF_WAVE);
    float* outBase = out + (size_t)b * ND_ * HW_ + (size_t)y * W_ + x0;

    // A fragments: 16-bit A 16x32 layout -> lane holds M=l16, K = kb*32 + lh*8 + [0..7]
    // in q[0] and kb*32 + 16 + lh*8 + [0..7] in q[1].
    Frag a[4];
    {
        const char* aBase = smemS + (size_t)(wv * SRC_PX + l16) * PIXSTRIDE * 2;
#pragma unroll
        for (int kb = 0; kb < 4; ++kb) {
            a[kb].q[0] = *(const uint4*)(aBase + kb * 64 + lh * 16);
            a[kb].q[1] = *(const uint4*)(aBase + kb * 64 + lh * 16 + 32);
        }
    }

    for (int dyi = 0; dyi < 9; ++dyi) {
        const int trow = wv + dyi;        // LDS tgt row for y + (dyi-4)

        // Load ALL B fragments for this dy up front (2 N-tiles x 4 K-chunks, 16 x
        // ds_load_b128 into distinct registers) so the scheduler can overlap LDS
        // latency with the WMMA chain instead of draining DScnt before every MMA.
        Frag bf[2][4];
#pragma unroll
        for (int nt = 0; nt < 2; ++nt) {
            const char* bBase = smemT +
                (size_t)(trow * TGT_PX + nt * 16 + l16) * PIXSTRIDE * 2 + lh * 16;
#pragma unroll
            for (int kb = 0; kb < 4; ++kb) {
                bf[nt][kb].q[0] = *(const uint4*)(bBase + kb * 64);
                bf[nt][kb].q[1] = *(const uint4*)(bBase + kb * 64 + 32);
            }
        }

#pragma unroll
        for (int nt = 0; nt < 2; ++nt) {
            v8f acc = {0.f, 0.f, 0.f, 0.f, 0.f, 0.f, 0.f, 0.f};
#pragma unroll
            for (int kb = 0; kb < 4; ++kb) {
                acc = __builtin_amdgcn_wmma_f32_16x16x32_bf16(
                    false, a[kb].v, false, bf[nt][kb].v, (short)0, acc, false, false);
            }
            // Dump D tile to LDS: lane owns (n = nt*16+l16, m = 8*lh + [0..7]) -> one
            // contiguous 8-float run per lane at [n][8*lh].
            float* dstn = dbuf + (nt * 16 + l16) * DSTR + 8 * lh;
            *(float2*)(dstn + 0) = make_float2(acc[0], acc[1]);
            *(float2*)(dstn + 2) = make_float2(acc[2], acc[3]);
            *(float2*)(dstn + 4) = make_float2(acc[4], acc[5]);
            *(float2*)(dstn + 6) = make_float2(acc[6], acc[7]);
        }

        // Coalesced write-out of the 9x16 diagonal band: each 16-lane half writes the
        // 16 consecutive x of one displacement channel (same-wave LDS ops are in-order,
        // so no barrier is needed between the dumps above and these reads).
#pragma unroll
        for (int pass = 0; pass < 5; ++pass) {
            int e = pass * 32 + lane;     // 0..159, 144 valid
            if (e < 144) {
                int dxg = e >> 4;         // 0..8  -> dx = dxg-4
                int m   = e & 15;         // src pixel within tile
                float v = dbuf[(m + dxg) * DSTR + m];   // n = m + dx + 4 = m + dxg
                int ch  = chidx(dyi - S_, dxg - S_);
                outBase[(size_t)ch * HW_ + m] = v;
            }
        }
    }
}

extern "C" void kernel_launch(void* const* d_in, const int* in_sizes, int n_in,
                              void* d_out, int out_size, void* d_ws, size_t ws_size,
                              hipStream_t stream) {
    const float* src = (const float*)d_in[0];
    const float* tgt = (const float*)d_in[1];
    float* out = (float*)d_out;

    dim3 grid(W_ / 16, H_ / 4, B_);   // 18 x 40 x 8
    dim3 block(128);                  // 4 wave32s, one output row each
    size_t shmem = (size_t)TGT_BYTES + SRC_BYTES + DBUF_BYTES;  // 128 KB of 320 KB WGP LDS
    hipLaunchKernelGGL(costvol_wmma_bf16, grid, block, shmem, stream, src, tgt, out);
}
